// CEASC_51024211476559
// MI455X (gfx1250) — matrix operand order repaired and checked by reading, not compiled, not run on hardware
//
#include <hip/hip_runtime.h>
#include <hip/hip_bf16.h>
#include <cstdint>

// ---------------------------------------------------------------------------
// CEASC head for MI455X (gfx1250): bf16 WMMA implicit-GEMM 3x3 convs.
// Halo-padded NHWC bf16 feature maps (no boundary selects, L2-resident),
// double-buffered LDS weight tiles (3 taps / tile), imm-offset A loads.
// ---------------------------------------------------------------------------

typedef __attribute__((ext_vector_type(16))) __bf16 v16bf;
typedef __attribute__((ext_vector_type(8)))  float  v8f;

union FragAB { v16bf v; int4 q[2]; };
union FragC  { v8f  v; float f[8]; };

constexpr int B_  = 4;
constexpr int C_  = 256;
constexpr int H_  = 128;
constexpr int W_  = 128;
constexpr int HW_ = H_ * W_;
constexpr int HP_ = H_ + 2;            // padded rows
constexpr int WP_ = W_ + 2;            // padded cols
constexpr int G_  = 32;                // groups
constexpr float EPS_ = 1e-5f;

// padded NHWC pixel base (h,w may be -1..128)
__device__ __forceinline__ size_t ppix(int b, int h, int w) {
  return ((size_t)(b * HP_ + h + 1) * WP_ + (w + 1)) * C_;
}

// ---------------------------------------------------------------------------
// Zero the padded feature buffers (halo must be 0 every call)
// ---------------------------------------------------------------------------
__global__ __launch_bounds__(256) void zero_ws_kernel(int4* __restrict__ p, size_t n4) {
  size_t i = (size_t)blockIdx.x * 256 + threadIdx.x;
  size_t stride = (size_t)gridDim.x * 256;
  const int4 z = make_int4(0, 0, 0, 0);
  for (; i < n4; i += stride) p[i] = z;
}

// ---------------------------------------------------------------------------
// Weight packing: fp32 [layer][co][ci][3][3] -> bf16 [br][layer][co][tap][ci]
// ---------------------------------------------------------------------------
__global__ __launch_bounds__(256) void pack_weights_kernel(
    const float* __restrict__ clsw, const float* __restrict__ regw,
    __bf16* __restrict__ wp) {
  size_t idx = (size_t)blockIdx.x * 256 + threadIdx.x;           // < 2*4*256*9*256
  int ci  = idx & 255;
  int tap = (idx >> 8) % 9;
  int co  = (int)((idx / (9 * 256)) & 255);
  int lay = (int)((idx / (9 * 256 * 256)) & 3);
  int br  = (int)( idx / ((size_t)9 * 256 * 256 * 4));
  const float* src = br ? regw : clsw;
  float v = src[(((size_t)lay * 256 + co) * 256 + ci) * 9 + tap];
  wp[idx] = (__bf16)v;
}

__global__ __launch_bounds__(256) void pack_head_kernel(
    const float* __restrict__ predw, __bf16* __restrict__ wph) {
  int idx = blockIdx.x * 256 + threadIdx.x;                      // < 16*9*256
  int ci  = idx & 255;
  int tap = (idx >> 8) % 9;
  int co  = idx / (9 * 256);
  float v = (co < 10) ? predw[((size_t)(co * 256 + ci)) * 9 + tap] : 0.0f;
  wph[idx] = (__bf16)v;
}

// x NCHW fp32 -> padded NHWC bf16 interior
__global__ __launch_bounds__(256) void convert_x_kernel(
    const float* __restrict__ x, __bf16* __restrict__ xbf) {
  size_t idx = (size_t)blockIdx.x * 256 + threadIdx.x;           // < B*HW*C
  int ci = idx & 255;
  size_t p = idx >> 8;
  int b  = (int)(p >> 14);
  int hw = (int)(p & (HW_ - 1));
  int h = hw >> 7, w = hw & 127;
  xbf[ppix(b, h, w) + ci] = (__bf16)x[((size_t)b * C_ + ci) * HW_ + hw];
}

// ---------------------------------------------------------------------------
// Global context: gc[b][c] = mean_{h,w} x[b][c]
// ---------------------------------------------------------------------------
__global__ __launch_bounds__(256) void gc_mean_kernel(
    const float* __restrict__ x, float* __restrict__ gc) {
  __shared__ float red[256];
  int bc = blockIdx.x;                                           // b*256+c
  const float* src = x + (size_t)bc * HW_;
  float s = 0.f;
  for (int i = threadIdx.x; i < HW_; i += 256) s += src[i];
  red[threadIdx.x] = s; __syncthreads();
  for (int st = 128; st > 0; st >>= 1) {
    if (threadIdx.x < st) red[threadIdx.x] += red[threadIdx.x + st];
    __syncthreads();
  }
  if (threadIdx.x == 0) gc[bc] = red[0] * (1.0f / HW_);
}

// g = W_global @ gc + b  for both branches: gvec[(br*B+b)*C + co]
__global__ __launch_bounds__(256) void gc_apply_kernel(
    const float* __restrict__ gc,
    const float* __restrict__ wcls, const float* __restrict__ bcls,
    const float* __restrict__ wreg, const float* __restrict__ breg,
    float* __restrict__ gvec) {
  int b  = blockIdx.x & 3;
  int br = blockIdx.x >> 2;
  int co = threadIdx.x;
  const float* w  = br ? wreg : wcls;
  const float* bb = br ? breg : bcls;
  float s = 0.f;
  for (int ci = 0; ci < C_; ++ci)
    s += gc[b * C_ + ci] * w[co * C_ + ci];
  gvec[((size_t)br * B_ + b) * C_ + co] = s + bb[co];
}

// ---------------------------------------------------------------------------
// AMM masks: 1x1 conv 256->2, softmax ch1, hard threshold
// ---------------------------------------------------------------------------
__global__ __launch_bounds__(256) void amm_kernel(
    const float* __restrict__ x,
    const float* __restrict__ wc, const float* __restrict__ bc,
    const float* __restrict__ wr, const float* __restrict__ brg,
    float* __restrict__ soft_cls, float* __restrict__ soft_reg,
    float* __restrict__ hard) {
  __shared__ float wl[4][256];   // [cls0, cls1, reg0, reg1]
  for (int i = threadIdx.x; i < 1024; i += 256) {
    int oc = i >> 8, ci = i & 255;
    wl[oc][ci] = (oc < 2) ? wc[oc * 256 + ci] : wr[(oc - 2) * 256 + ci];
  }
  __syncthreads();
  int p  = blockIdx.x * 256 + threadIdx.x;                       // < B*HW
  int b  = p >> 14;
  int hw = p & (HW_ - 1);
  float a0 = 0.f, a1 = 0.f, a2 = 0.f, a3 = 0.f;
  const float* src = x + (size_t)b * C_ * HW_ + hw;
  for (int ci = 0; ci < C_; ++ci) {
    float xv = src[(size_t)ci * HW_];
    a0 += xv * wl[0][ci]; a1 += xv * wl[1][ci];
    a2 += xv * wl[2][ci]; a3 += xv * wl[3][ci];
  }
  float sc = 1.f / (1.f + __expf((a0 + bc[0]) - (a1 + bc[1])));
  float sr = 1.f / (1.f + __expf((a2 + brg[0]) - (a3 + brg[1])));
  soft_cls[p] = sc;
  soft_reg[p] = sr;
  hard[p]            = sc > 0.5f ? 1.f : 0.f;
  hard[B_ * HW_ + p] = sr > 0.5f ? 1.f : 0.f;
}

// ---------------------------------------------------------------------------
// Main CESC conv: 3x3, 256->256, implicit GEMM with v_wmma_f32_16x16x32_bf16.
// Block = 8 waves (4M x 2N): 128 pixels (one row) x 64 out-channels.
// Grid = (B*H, C/64). 24 iterations: 8 ci-chunks x 3 tap-groups (dy rows).
// Double-buffered [3tap][64co][32ci] LDS tile (2 x 12KB), 12 WMMAs/barrier.
// Padded input -> no boundary logic; A loads use immediate dx offsets.
// Epilogue: +bias, hard-mask gating with global fill, write fp32 NHWC ybuf.
// ---------------------------------------------------------------------------
__global__ __launch_bounds__(256) void conv3x3_cesc_kernel(
    const __bf16* __restrict__ fin,   // padded NHWC bf16 [B*130*130][256]
    const __bf16* __restrict__ wp,    // [256 co][9 tap][256 ci]
    const float*  __restrict__ bias,  // [256]
    const float*  __restrict__ maskh, // [B*HW]
    const float*  __restrict__ gvec,  // [B*256]
    float*        __restrict__ ybuf)  // dense NHWC fp32 [B*HW][256]
{
  __shared__ __bf16 wt[2][3 * 64 * 32];   // 2 x 12KB: [tapin][co][ci]
  const int tid   = threadIdx.x;
  const int lane  = tid & 31;
  const int wid   = tid >> 5;
  const int b     = blockIdx.x >> 7;
  const int h     = blockIdx.x & 127;
  const int coBlk = blockIdx.y << 6;
  const int waveM = wid & 3;
  const int waveN = wid >> 2;
  const int pxbase = waveM << 5;                 // 32 pixels per M-wave
  const int coW    = coBlk + (waveN << 5);       // 32 channels per N-wave
  const int m     = lane & 15;
  const int hi    = lane >> 4;
  const int abase = hi << 3;                     // A-frag K base (ISA 16-bit A 16x32)
  const int kb    = hi << 3;                     // B-frag K base (same k order as A)
  const int c0    = (waveN << 5) + m;            // B-frag co base within 64-block

  FragC acc[2][2] = {};

  // stage tap-group g, ci-chunk ci0 into buffer buf (6 x b128 per thread)
  auto stageW = [&](int g, int ci0, int buf) {
#pragma unroll
    for (int i = 0; i < 6; ++i) {
      int f = tid + (i << 8);                    // < 1536 int4
      int rowid = f >> 2;                        // (tapin*64 + co)
      int tapin = rowid >> 6;
      int co    = rowid & 63;
      int seg   = f & 3;
      const int4* src = (const int4*)(wp +
          (((size_t)(coBlk + co) * 9 + g * 3 + tapin) * 256 + ci0 + seg * 8));
      ((int4*)wt[buf])[f] = *src;
    }
  };

  stageW(0, 0, 0);
  __syncthreads();

  int g = 0, ci0 = 0, parity = 0;
  for (int itr = 0; itr < 24; ++itr) {
    int ng = g + 1, nci = ci0;
    if (ng == 3) { ng = 0; nci += 32; }
    if (itr < 23) stageW(ng, nci, parity ^ 1);   // overlaps WMMA burst

    const __bf16* wtc = wt[parity];
    // A base per M-tile for this dy row (h + g - 1); padded => always valid
#pragma unroll
    for (int mt = 0; mt < 2; ++mt) {
      const __bf16* ab = fin +
          (ppix(b, h + g - 1, pxbase + (mt << 4) + m) + ci0 + abase);
#pragma unroll
      for (int tapin = 0; tapin < 3; ++tapin) {
        const int dxo = (tapin - 1) * C_;        // dx in {-1,0,+1} -> +/-512B imm
        FragAB a;
        a.q[0] = *(const int4*)(ab + dxo);
        a.q[1] = *(const int4*)(ab + dxo + 16);
#pragma unroll
        for (int nt = 0; nt < 2; ++nt) {
          FragAB bf;
          const __bf16* brp = wtc + ((tapin << 6) + c0 + (nt << 4)) * 32 + kb;
          bf.q[0] = *(const int4*)brp;
          bf.q[1] = *(const int4*)(brp + 16);
          acc[mt][nt].v = __builtin_amdgcn_wmma_f32_16x16x32_bf16(
              false, a.v, false, bf.v, (short)0, acc[mt][nt].v, false, false);
        }
      }
    }
    __syncthreads();
    g = ng; ci0 = nci; parity ^= 1;
  }

  // Epilogue: bias, mask gating (y*m + g*(1-m)), store fp32 dense NHWC
  float bco[2], gco[2];
#pragma unroll
  for (int nt = 0; nt < 2; ++nt) {
    int co = coW + (nt << 4) + m;
    bco[nt] = bias[co];
    gco[nt] = gvec[b * C_ + co];
  }
#pragma unroll
  for (int mt = 0; mt < 2; ++mt) {
#pragma unroll
    for (int r = 0; r < 8; ++r) {
      int px = pxbase + (mt << 4) + (hi << 3) + r;               // D layout: M = r + 8*hi
      float mk = maskh[((size_t)b << 14) + (h << 7) + px];
      size_t orow = (((size_t)(b * H_ + h)) * W_ + px) * C_;
#pragma unroll
      for (int nt = 0; nt < 2; ++nt) {
        float v = acc[mt][nt].f[r] + bco[nt];
        v = v * mk + gco[nt] * (1.0f - mk);
        ybuf[orow + coW + (nt << 4) + m] = v;
      }
    }
  }
}

// ---------------------------------------------------------------------------
// GroupNorm: stats (mean, rstd) then apply (+gamma,beta,ReLU) -> padded bf16
// ---------------------------------------------------------------------------
__global__ __launch_bounds__(256) void gn_stats_kernel(
    const float* __restrict__ ybuf, float* __restrict__ gst) {
  __shared__ float rs[256], rq[256];
  int b = blockIdx.x >> 5, g = blockIdx.x & 31;
  float s = 0.f, q = 0.f;
  for (int pix = threadIdx.x; pix < HW_; pix += 256) {
    const float4* pp = (const float4*)(ybuf + ((size_t)b * HW_ + pix) * C_ + g * 8);
    float4 u = pp[0], v = pp[1];
    s += u.x + u.y + u.z + u.w + v.x + v.y + v.z + v.w;
    q += u.x*u.x + u.y*u.y + u.z*u.z + u.w*u.w +
         v.x*v.x + v.y*v.y + v.z*v.z + v.w*v.w;
  }
  rs[threadIdx.x] = s; rq[threadIdx.x] = q; __syncthreads();
  for (int st = 128; st > 0; st >>= 1) {
    if (threadIdx.x < st) { rs[threadIdx.x] += rs[threadIdx.x + st];
                            rq[threadIdx.x] += rq[threadIdx.x + st]; }
    __syncthreads();
  }
  if (threadIdx.x == 0) {
    const float n = (float)HW_ * 8.0f;
    float mean = rs[0] / n;
    float var  = rq[0] / n - mean * mean;
    gst[2 * blockIdx.x]     = mean;
    gst[2 * blockIdx.x + 1] = rsqrtf(var + EPS_);
  }
}

__global__ __launch_bounds__(256) void gn_apply_kernel(
    const float* __restrict__ ybuf, const float* __restrict__ gst,
    const float* __restrict__ gamma, const float* __restrict__ beta,
    __bf16* __restrict__ fout) {
  size_t idx = (size_t)blockIdx.x * 256 + threadIdx.x;           // < B*HW*C
  int c = idx & 255;
  size_t p = idx >> 8;
  int b  = (int)(p >> 14);
  int hw = (int)(p & (HW_ - 1));
  int g = c >> 3;
  float mean = gst[2 * (b * G_ + g)];
  float rstd = gst[2 * (b * G_ + g) + 1];
  float v = (ybuf[idx] - mean) * rstd * gamma[c] + beta[c];
  fout[ppix(b, hw >> 7, hw & 127) + c] = (__bf16)fmaxf(v, 0.0f);
}

// ---------------------------------------------------------------------------
// cls head: 3x3 conv 256->10 (padded to 16 co) with WMMA; writes NCHW fp32
// ---------------------------------------------------------------------------
__global__ __launch_bounds__(256) void conv3x3_head_kernel(
    const __bf16* __restrict__ fin, const __bf16* __restrict__ wph,
    const float* __restrict__ pb, float* __restrict__ outp) {
  __shared__ __bf16 wt[2][3 * 16 * 32];   // 2 x 3KB: [tapin][co][ci]
  const int tid  = threadIdx.x;
  const int lane = tid & 31;
  const int wid  = tid >> 5;
  const int b = blockIdx.x >> 7, h = blockIdx.x & 127;
  const int pxbase = wid << 4;
  const int m  = lane & 15;
  const int hi = lane >> 4;
  const int abase = hi << 3;
  const int kb    = hi << 3;
  FragC acc = {};

  auto stageW = [&](int g, int ci0, int buf) {
    if (tid < 192) {                              // 192 x b128
      int rowid = tid >> 2;                       // (tapin*16 + co)
      int tapin = rowid >> 4;
      int co    = rowid & 15;
      int seg   = tid & 3;
      const int4* src = (const int4*)(wph +
          (((size_t)co * 9 + g * 3 + tapin) * 256 + ci0 + seg * 8));
      ((int4*)wt[buf])[tid] = *src;
    }
  };

  stageW(0, 0, 0);
  __syncthreads();

  int g = 0, ci0 = 0, parity = 0;
  for (int itr = 0; itr < 24; ++itr) {
    int ng = g + 1, nci = ci0;
    if (ng == 3) { ng = 0; nci += 32; }
    if (itr < 23) stageW(ng, nci, parity ^ 1);

    const __bf16* wtc = wt[parity];
    const __bf16* ab = fin + (ppix(b, h + g - 1, pxbase + m) + ci0 + abase);
#pragma unroll
    for (int tapin = 0; tapin < 3; ++tapin) {
      const int dxo = (tapin - 1) * C_;
      FragAB a, bf;
      a.q[0] = *(const int4*)(ab + dxo);
      a.q[1] = *(const int4*)(ab + dxo + 16);
      const __bf16* brp = wtc + ((tapin << 4) + m) * 32 + kb;
      bf.q[0] = *(const int4*)brp;
      bf.q[1] = *(const int4*)(brp + 16);
      acc.v = __builtin_amdgcn_wmma_f32_16x16x32_bf16(
          false, a.v, false, bf.v, (short)0, acc.v, false, false);
    }
    __syncthreads();
    g = ng; ci0 = nci; parity ^= 1;
  }
  const int co = m;
  if (co < 10) {
    float bb = pb[co];
#pragma unroll
    for (int r = 0; r < 8; ++r) {
      int px = pxbase + (hi << 3) + r;
      outp[(((size_t)b * 10 + co) * H_ + h) * W_ + px] = acc.f[r] + bb;
    }
  }
}

// reg head: 1x1 conv 256->4, NCHW fp32 output (reads padded NHWC)
__global__ __launch_bounds__(256) void reg_head_kernel(
    const __bf16* __restrict__ fin, const float* __restrict__ w,
    const float* __restrict__ pb, float* __restrict__ outp) {
  __shared__ float wl[4 * 256];
  for (int i = threadIdx.x; i < 1024; i += 256) wl[i] = w[i];
  __syncthreads();
  int p  = blockIdx.x * 256 + threadIdx.x;                        // < B*HW
  int b  = p >> 14;
  int hw = p & (HW_ - 1);
  const __bf16* src = fin + ppix(b, hw >> 7, hw & 127);
  float a0 = 0.f, a1 = 0.f, a2 = 0.f, a3 = 0.f;
  for (int ci = 0; ci < C_; ++ci) {
    float fv = (float)src[ci];
    a0 += fv * wl[ci]; a1 += fv * wl[256 + ci];
    a2 += fv * wl[512 + ci]; a3 += fv * wl[768 + ci];
  }
  outp[((size_t)b * 4 + 0) * HW_ + hw] = a0 + pb[0];
  outp[((size_t)b * 4 + 1) * HW_ + hw] = a1 + pb[1];
  outp[((size_t)b * 4 + 2) * HW_ + hw] = a2 + pb[2];
  outp[((size_t)b * 4 + 3) * HW_ + hw] = a3 + pb[3];
}

// ---------------------------------------------------------------------------
// Host launch
// ---------------------------------------------------------------------------
extern "C" void kernel_launch(void* const* d_in, const int* in_sizes, int n_in,
                              void* d_out, int out_size, void* d_ws, size_t ws_size,
                              hipStream_t stream) {
  const float* x        = (const float*)d_in[0];
  const float* cammw    = (const float*)d_in[1];
  const float* cammb    = (const float*)d_in[2];
  const float* rammw    = (const float*)d_in[3];
  const float* rammb    = (const float*)d_in[4];
  const float* cconvw   = (const float*)d_in[5];
  const float* cconvb   = (const float*)d_in[6];
  const float* cgng     = (const float*)d_in[7];
  const float* cgnb     = (const float*)d_in[8];
  const float* rconvw   = (const float*)d_in[9];
  const float* rconvb   = (const float*)d_in[10];
  const float* rgng     = (const float*)d_in[11];
  const float* rgnb     = (const float*)d_in[12];
  const float* gclsw    = (const float*)d_in[13];
  const float* gclsb    = (const float*)d_in[14];
  const float* gregw    = (const float*)d_in[15];
  const float* gregb    = (const float*)d_in[16];
  const float* predw    = (const float*)d_in[17];
  const float* predb    = (const float*)d_in[18];
  const float* rpredw   = (const float*)d_in[19];
  const float* rpredb   = (const float*)d_in[20];
  float* dout = (float*)d_out;

  constexpr size_t FEAT_E  = (size_t)B_ * HW_ * C_;               // dense elements
  constexpr size_t FEATP_E = (size_t)B_ * HP_ * WP_ * C_;         // padded elements
  constexpr size_t FEATP_B = (FEATP_E * 2 + 255) & ~(size_t)255;  // bytes, aligned
  constexpr size_t OFF_XBF  = 0;
  constexpr size_t OFF_PING = OFF_XBF  + FEATP_B;
  constexpr size_t OFF_PONG = OFF_PING + FEATP_B;
  constexpr size_t OFF_YBUF = OFF_PONG + FEATP_B;
  constexpr size_t OFF_WP   = OFF_YBUF + FEAT_E * 4;
  constexpr size_t OFF_WPH  = OFF_WP   + (size_t)8 * 256 * 9 * 256 * 2;
  constexpr size_t OFF_GC   = OFF_WPH  + (size_t)16 * 9 * 256 * 2;
  constexpr size_t OFF_GVEC = OFF_GC   + 1024 * 4;
  constexpr size_t OFF_MASK = OFF_GVEC + 2048 * 4;
  constexpr size_t OFF_GST  = OFF_MASK + (size_t)2 * B_ * HW_ * 4;

  char* ws = (char*)d_ws;
  __bf16* xbf  = (__bf16*)(ws + OFF_XBF);
  __bf16* ping = (__bf16*)(ws + OFF_PING);
  __bf16* pong = (__bf16*)(ws + OFF_PONG);
  float*  ybuf = (float*) (ws + OFF_YBUF);
  __bf16* wp   = (__bf16*)(ws + OFF_WP);
  __bf16* wph  = (__bf16*)(ws + OFF_WPH);
  float*  gc   = (float*) (ws + OFF_GC);
  float*  gvec = (float*) (ws + OFF_GVEC);
  float*  mskh = (float*) (ws + OFF_MASK);
  float*  gst  = (float*) (ws + OFF_GST);

  // d_out layout: cls_out[4,10,128,128] | reg_out[4,4,128,128] | soft_cls | soft_reg
  float* out_cls  = dout;
  float* out_reg  = dout + (size_t)B_ * 10 * HW_;
  float* soft_cls = out_reg + (size_t)B_ * 4 * HW_;
  float* soft_reg = soft_cls + (size_t)B_ * HW_;

  // zero the 3 padded feature buffers (provides the zero halo)
  zero_ws_kernel<<<4096, 256, 0, stream>>>((int4*)ws, (3 * FEATP_B) / 16);

  pack_weights_kernel<<<18432, 256, 0, stream>>>(cconvw, rconvw, wp);
  pack_head_kernel<<<144, 256, 0, stream>>>(predw, wph);
  convert_x_kernel<<<65536, 256, 0, stream>>>(x, xbf);
  gc_mean_kernel<<<1024, 256, 0, stream>>>(x, gc);
  gc_apply_kernel<<<8, 256, 0, stream>>>(gc, gclsw, gclsb, gregw, gregb, gvec);
  amm_kernel<<<256, 256, 0, stream>>>(x, cammw, cammb, rammw, rammb,
                                      soft_cls, soft_reg, mskh);

  for (int br = 0; br < 2; ++br) {
    const float* convb = br ? rconvb : cconvb;
    const float* gng   = br ? rgng   : cgng;
    const float* gnb   = br ? rgnb   : cgnb;
    const __bf16* fi = xbf;
    __bf16* fo = ping;
    for (int i = 0; i < 4; ++i) {
      conv3x3_cesc_kernel<<<dim3(512, 4), 256, 0, stream>>>(
          fi, wp + ((size_t)(br * 4 + i)) * 256 * 9 * 256,
          convb + i * 256, mskh + (size_t)br * B_ * HW_,
          gvec + (size_t)br * B_ * C_, ybuf);
      gn_stats_kernel<<<128, 256, 0, stream>>>(ybuf, gst);
      gn_apply_kernel<<<65536, 256, 0, stream>>>(ybuf, gst,
          gng + i * 256, gnb + i * 256, fo);
      fi = fo;
      fo = (fo == ping) ? pong : ping;
    }
    if (br == 0)
      conv3x3_head_kernel<<<512, 256, 0, stream>>>(fi, wph, predb, out_cls);
    else
      reg_head_kernel<<<256, 256, 0, stream>>>(fi, rpredw, rpredb, out_reg);
  }
}
